// SelfAttention_63720134804171
// MI455X (gfx1250) — matrix operand order repaired and checked by reading, not compiled, not run on hardware
//
#include <hip/hip_runtime.h>
#include <hip/hip_bf16.h>

#define NBATCH 2
#define HEADS  16
#define SEQ    2048
#define HD     64
#define EMB    1024

typedef _Float16 v16h __attribute__((ext_vector_type(16)));
typedef _Float16 v8h  __attribute__((ext_vector_type(8)));
typedef float    v8f  __attribute__((ext_vector_type(8)));

__device__ __forceinline__ v8f wmma16(v16h a, v16h b, v8f c) {
  // D = A(16x32 f16) * B(32x16 f16) + C(16x16 f32)
  return __builtin_amdgcn_wmma_f32_16x16x32_f16(false, a, false, b, (short)0, c,
                                                false, false);
}

// ---- CDNA5 async copy: global -> LDS, ASYNCcnt-tracked ------------------
// Each lane copies 16B; the LDS address is the lane's VGPR value plus the offset (ISA 10 / 15.18.3).
__device__ __forceinline__ void async_ld_b128(_Float16* lds_dst,
                                              const _Float16* gsrc) {
  unsigned int l = (unsigned int)(unsigned long long)lds_dst;
  asm volatile("global_load_async_to_lds_b128 %0, %1, off" ::"v"(l), "v"(gsrc)
               : "memory");
}
__device__ __forceinline__ void wait_async0() {
  asm volatile("s_wait_asynccnt 0x0" ::: "memory");
}

// ---- operand loaders (per CDNA5 ISA 7.12.2 VGPR layouts) ----------------
// A tile 16x32 from row-major f16 matrix, row stride ld.
// lanes 0-15 hold row M=lane: halves[0..7]=K 0..7, halves[8..15]=K 16..23
// lanes 16-31 hold row M=lane-16: halves[0..7]=K 8..15, [8..15]=K 24..31
__device__ __forceinline__ v16h load_A16(const _Float16* base, int ld, int m0,
                                         int k0, int lane) {
  const _Float16* p =
      base + (size_t)(m0 + (lane & 15)) * ld + k0 + ((lane >> 4) << 3);
  v8h lo = *(const v8h*)p;
  v8h hi = *(const v8h*)(p + 16);
  v16h r;
#pragma unroll
  for (int i = 0; i < 8; ++i) { r[i] = lo[i]; r[8 + i] = hi[i]; }
  return r;
}

// Same A tile but sourced from f32 row-major, converted to f16.
__device__ __forceinline__ v16h load_A32(const float* base, int ld, int m0,
                                         int k0, int lane) {
  const float* p =
      base + (size_t)(m0 + (lane & 15)) * ld + k0 + ((lane >> 4) << 3);
  v16h r;
#pragma unroll
  for (int i = 0; i < 8; ++i) {
    r[i]     = (_Float16)p[i];
    r[8 + i] = (_Float16)p[i + 16];
  }
  return r;
}

// B tile 32x16 where columns are contiguous in memory:
// element(k, c) = base[(c0+c)*ld + k0 + k].
// lanes 0-15 hold K=0..7,16..23 of column c0+lane; lanes 16-31 hold K=8..15,24..31.
__device__ __forceinline__ v16h load_B16(const _Float16* base, int ld, int c0,
                                         int k0, int lane) {
  const _Float16* p =
      base + (size_t)(c0 + (lane & 15)) * ld + k0 + ((lane >> 4) << 3);
  v8h lo = *(const v8h*)p;
  v8h hi = *(const v8h*)(p + 16);
  v16h r;
#pragma unroll
  for (int i = 0; i < 8; ++i) { r[i] = lo[i]; r[8 + i] = hi[i]; }
  return r;
}

// ---- kernel 0: weight f32 -> f16 conversion -----------------------------
__global__ void prep_weights(const float* __restrict__ wq,
                             const float* __restrict__ wk,
                             const float* __restrict__ wv,
                             const float* __restrict__ wo,
                             _Float16* __restrict__ wqf,
                             _Float16* __restrict__ wkf,
                             _Float16* __restrict__ wvf,
                             _Float16* __restrict__ wof) {
  int i = blockIdx.x * 256 + threadIdx.x;
  if (i < HD * HD) {
    wqf[i] = (_Float16)wq[i];
    wkf[i] = (_Float16)wk[i];
    wvf[i] = (_Float16)wv[i];
  }
  if (i < EMB * EMB) wof[i] = (_Float16)wo[i];
}

// ---- kernel 1: per-head QKV projection ----------------------------------
// out_j = sum_k x_k * W[j][k] + b_j  (x @ W.T). W rows are K-contiguous ->
// W itself is the column-major B operand.
// Q,K stored [head][L][64]; V stored transposed [head][64][L].
__global__ void __launch_bounds__(128)
qkv_kernel(const float* __restrict__ q_in, const float* __restrict__ k_in,
           const float* __restrict__ v_in, const _Float16* __restrict__ wqf,
           const _Float16* __restrict__ wkf, const _Float16* __restrict__ wvf,
           const float* __restrict__ bq, const float* __restrict__ bk,
           const float* __restrict__ bv, _Float16* __restrict__ Qp,
           _Float16* __restrict__ Kp, _Float16* __restrict__ Vt) {
  int lane = threadIdx.x & 31, warp = threadIdx.x >> 5;
  int head = blockIdx.y;  // n*HEADS + h
  int n = head >> 4, h = head & 15;
  int l0 = blockIdx.x * 64 + warp * 16;
  int c = lane & 15, rofs = (lane >> 4) << 3;

  const float* xq = q_in + (size_t)(n * SEQ + l0) * EMB + h * HD;
  const float* xk = k_in + (size_t)(n * SEQ + l0) * EMB + h * HD;
  const float* xv = v_in + (size_t)(n * SEQ + l0) * EMB + h * HD;

  {  // Q
    v16h a0 = load_A32(xq, EMB, 0, 0, lane);
    v16h a1 = load_A32(xq, EMB, 0, 32, lane);
#pragma unroll
    for (int t = 0; t < 4; ++t) {
      v8f acc = {};
      acc = wmma16(a0, load_B16(wqf, HD, t * 16, 0, lane), acc);
      acc = wmma16(a1, load_B16(wqf, HD, t * 16, 32, lane), acc);
      float bias = bq[t * 16 + c];
#pragma unroll
      for (int v = 0; v < 8; ++v)
        Qp[((size_t)head * SEQ + l0 + v + rofs) * HD + t * 16 + c] =
            (_Float16)(acc[v] + bias);
    }
  }
  {  // K
    v16h a0 = load_A32(xk, EMB, 0, 0, lane);
    v16h a1 = load_A32(xk, EMB, 0, 32, lane);
#pragma unroll
    for (int t = 0; t < 4; ++t) {
      v8f acc = {};
      acc = wmma16(a0, load_B16(wkf, HD, t * 16, 0, lane), acc);
      acc = wmma16(a1, load_B16(wkf, HD, t * 16, 32, lane), acc);
      float bias = bk[t * 16 + c];
#pragma unroll
      for (int v = 0; v < 8; ++v)
        Kp[((size_t)head * SEQ + l0 + v + rofs) * HD + t * 16 + c] =
            (_Float16)(acc[v] + bias);
    }
  }
  {  // V, stored transposed so the PV B-operand tile is row-gatherable
    v16h a0 = load_A32(xv, EMB, 0, 0, lane);
    v16h a1 = load_A32(xv, EMB, 0, 32, lane);
#pragma unroll
    for (int t = 0; t < 4; ++t) {
      v8f acc = {};
      acc = wmma16(a0, load_B16(wvf, HD, t * 16, 0, lane), acc);
      acc = wmma16(a1, load_B16(wvf, HD, t * 16, 32, lane), acc);
      float bias = bv[t * 16 + c];
#pragma unroll
      for (int v = 0; v < 8; ++v)
        Vt[((size_t)head * HD + t * 16 + c) * SEQ + l0 + v + rofs] =
            (_Float16)(acc[v] + bias);
    }
  }
}

// ---- kernel 2: flash attention with async-to-LDS K/V staging ------------
// 8 waves/block share one head; K (32x64) and V (64x32) tiles are DMA'd
// into LDS with global_load_async_to_lds_b128, double-buffered so the next
// tile's copy overlaps the current tile's WMMA + online softmax.
__global__ void __launch_bounds__(256)
attn_kernel(const _Float16* __restrict__ Qp, const _Float16* __restrict__ Kp,
            const _Float16* __restrict__ Vt, const int* __restrict__ mask,
            _Float16* __restrict__ Ao) {
  __shared__ _Float16 Kt[2][32 * HD];     // [seq 32][d 64]
  __shared__ _Float16 Vl[2][HD * 32];     // [d 64][seq 32]
  __shared__ _Float16 Plds[8][16][40];

  int lane = threadIdx.x & 31, warp = threadIdx.x >> 5;
  int head = blockIdx.y;
  int n = head >> 4, h = head & 15;
  int q0 = (blockIdx.x * 8 + warp) * 16;

  const _Float16* Qb = Qp + (size_t)head * SEQ * HD;
  const _Float16* Kb = Kp + (size_t)head * SEQ * HD;
  const _Float16* Vb = Vt + (size_t)head * HD * SEQ;
  const int* mb = mask + (size_t)n * SEQ * SEQ;

  v16h aq0 = load_A16(Qb, HD, q0, 0, lane);
  v16h aq1 = load_A16(Qb, HD, q0, 32, lane);

  v8f o[4] = {{}, {}, {}, {}};
  float mi[8], li[8];
#pragma unroll
  for (int v = 0; v < 8; ++v) { mi[v] = -3.0e38f; li[v] = 0.0f; }

  int cA = lane & 15;
  int rofs = (lane >> 4) << 3;
  const float sc = 0.03125f;  // 1/sqrt(EMBED)

  // copy partition: wave w moves 512B of K and 512B of V per tile
  int kOfs = warp * 256 + lane * 8;                    // elements into K tile
  int vRow = warp * 8 + (lane >> 2);                   // d-row of V tile
  int vCol = (lane & 3) * 8;                           // elements within row

  // prologue: DMA tile 0 into buffer 0
  async_ld_b128(&Kt[0][kOfs], Kb + kOfs);
  async_ld_b128(&Vl[0][vRow * 32 + vCol], Vb + (size_t)vRow * SEQ + vCol);

  for (int j0 = 0; j0 < SEQ; j0 += 32) {
    int cur = (j0 >> 5) & 1;
    wait_async0();      // own DMA for buf[cur] complete
    __syncthreads();    // everyone's DMA for buf[cur] complete
    if (j0 + 32 < SEQ) {
      int nb = cur ^ 1, nj = j0 + 32;
      async_ld_b128(&Kt[nb][kOfs], Kb + (size_t)nj * HD + kOfs);
      async_ld_b128(&Vl[nb][vRow * 32 + vCol],
                    Vb + (size_t)vRow * SEQ + nj + vCol);
    }
    __builtin_prefetch(mb + (size_t)q0 * SEQ + j0, 0, 0);  // global_prefetch_b8

    const _Float16* Kls = &Kt[cur][0];
    const _Float16* Vls = &Vl[cur][0];

    // S = Q * K^T : WMMA K-dim is d (64), B columns are seq positions.
    v8f s0 = {}, s1 = {};
    s0 = wmma16(aq0, load_B16(Kls, HD, 0, 0, lane), s0);
    s0 = wmma16(aq1, load_B16(Kls, HD, 0, 32, lane), s0);
    s1 = wmma16(aq0, load_B16(Kls, HD, 16, 0, lane), s1);
    s1 = wmma16(aq1, load_B16(Kls, HD, 16, 32, lane), s1);

#pragma unroll
    for (int v = 0; v < 8; ++v) {
      int row = q0 + v + rofs;
      float a = s0[v] * sc;
      float b = s1[v] * sc;
      if (mb[(size_t)row * SEQ + j0 + cA] == 0) a = -1.0e20f;
      if (mb[(size_t)row * SEQ + j0 + 16 + cA] == 0) b = -1.0e20f;
      float rmax = fmaxf(a, b);
#pragma unroll
      for (int off = 8; off; off >>= 1)
        rmax = fmaxf(rmax, __shfl_xor(rmax, off, 32));
      float mnew = fmaxf(mi[v], rmax);
      float corr = __expf(mi[v] - mnew);
      float pa = __expf(a - mnew);
      float pb = __expf(b - mnew);
      float rsum = pa + pb;
#pragma unroll
      for (int off = 8; off; off >>= 1) rsum += __shfl_xor(rsum, off, 32);
      li[v] = li[v] * corr + rsum;
      mi[v] = mnew;
#pragma unroll
      for (int t = 0; t < 4; ++t) o[t][v] = o[t][v] * corr;
      Plds[warp][v + rofs][cA]      = (_Float16)pa;
      Plds[warp][v + rofs][cA + 16] = (_Float16)pb;
    }

    // Re-read P in A layout (intra-wave LDS ops are in order);
    // O += P(16x32) * V(32x64)
    v16h ap = load_A16(&Plds[warp][0][0], 40, 0, 0, lane);
#pragma unroll
    for (int t = 0; t < 4; ++t)
      o[t] = wmma16(ap, load_B16(Vls, 32, t * 16, 0, lane), o[t]);
  }

#pragma unroll
  for (int v = 0; v < 8; ++v) {
    float inv = 1.0f / li[v];
    int row = q0 + v + rofs;
    size_t ob = (size_t)(n * SEQ + row) * EMB + h * HD;
#pragma unroll
    for (int t = 0; t < 4; ++t)
      Ao[ob + t * 16 + cA] = (_Float16)(o[t][v] * inv);
  }
}

// ---- kernel 3: output projection [N*L,1024] @ wo.T + bo (f32 out) -------
__global__ void __launch_bounds__(128)
proj_kernel(const _Float16* __restrict__ Ao, const _Float16* __restrict__ wof,
            const float* __restrict__ bo, float* __restrict__ out) {
  int lane = threadIdx.x & 31, warp = threadIdx.x >> 5;
  int col0 = blockIdx.x * 16;
  int row0 = (blockIdx.y * 4 + warp) * 16;
  v8f acc = {};
  for (int k0 = 0; k0 < EMB; k0 += 64) {
    acc = wmma16(load_A16(Ao, EMB, row0, k0, lane),
                 load_B16(wof, EMB, col0, k0, lane), acc);
    acc = wmma16(load_A16(Ao, EMB, row0, k0 + 32, lane),
                 load_B16(wof, EMB, col0, k0 + 32, lane), acc);
  }
  int c = lane & 15, rofs = (lane >> 4) << 3;
  float bias = bo[col0 + c];
#pragma unroll
  for (int v = 0; v < 8; ++v)
    out[(size_t)(row0 + v + rofs) * EMB + col0 + c] = acc[v] + bias;
}

// ---- host launch ---------------------------------------------------------
extern "C" void kernel_launch(void* const* d_in, const int* in_sizes, int n_in,
                              void* d_out, int out_size, void* d_ws,
                              size_t ws_size, hipStream_t stream) {
  (void)in_sizes; (void)n_in; (void)out_size; (void)ws_size;
  const float* value = (const float*)d_in[0];
  const float* key   = (const float*)d_in[1];
  const float* query = (const float*)d_in[2];
  const int*   mask  = (const int*)d_in[3];
  const float* wq = (const float*)d_in[4];
  const float* bq = (const float*)d_in[5];
  const float* wk = (const float*)d_in[6];
  const float* bk = (const float*)d_in[7];
  const float* wv = (const float*)d_in[8];
  const float* bv = (const float*)d_in[9];
  const float* wo = (const float*)d_in[10];
  const float* bo = (const float*)d_in[11];
  float* out = (float*)d_out;

  const size_t PH = (size_t)NBATCH * HEADS * SEQ * HD;  // 4 Mi f16 each
  _Float16* ws  = (_Float16*)d_ws;
  _Float16* Qp  = ws;
  _Float16* Kp  = Qp + PH;
  _Float16* Vt  = Kp + PH;
  _Float16* Ao  = Vt + PH;
  _Float16* wof = Ao + PH;
  _Float16* wqf = wof + (size_t)EMB * EMB;
  _Float16* wkf = wqf + HD * HD;
  _Float16* wvf = wkf + HD * HD;

  hipLaunchKernelGGL(prep_weights, dim3((EMB * EMB + 255) / 256), dim3(256), 0,
                     stream, wq, wk, wv, wo, wqf, wkf, wvf, wof);

  hipLaunchKernelGGL(qkv_kernel, dim3(SEQ / 64, NBATCH * HEADS), dim3(128), 0,
                     stream, query, key, value, wqf, wkf, wvf, bq, bk, bv, Qp,
                     Kp, Vt);

  hipLaunchKernelGGL(attn_kernel, dim3(SEQ / 128, NBATCH * HEADS), dim3(256), 0,
                     stream, Qp, Kp, Vt, mask, Ao);

  hipLaunchKernelGGL(proj_kernel, dim3(EMB / 16, (NBATCH * SEQ) / 64),
                     dim3(128), 0, stream, Ao, wof, bo, out);
}
